// QKVAttention_1331439862273
// MI455X (gfx1250) — matrix operand order repaired and checked by reading, compiled, not run on hardware
//
#include <hip/hip_runtime.h>
#include <hip/hip_bf16.h>

typedef __attribute__((ext_vector_type(16))) _Float16 v16h;
typedef __attribute__((ext_vector_type(8)))  _Float16 v8h;
typedef __attribute__((ext_vector_type(4)))  _Float16 v4h;
typedef __attribute__((ext_vector_type(8)))  float    v8f;

#define TSEQ   2048
#define CHD    64
#define HEADS  8
#define NBATCH 4
#define NHTOT  (NBATCH*HEADS)  // 32 batch*head pairs
#define BM     64              // query rows per block
#define BN     128             // keys per tile
#define NSUB   (BN/16)         // 8 key subtiles
#define TSTR   72              // LDS transpose-tile stride (pre-pass)
#define PSTR   136             // LDS stride for P rows (272B, 16B-aligned)
#define OSTR   68              // padded f32 stride for output staging
#define LOG2E  1.44269504088896340736f

// Build a 16-half fragment from two 16-byte chunks (b128 loads, global or LDS).
__device__ __forceinline__ v16h ld_frag(const _Float16* p0, const _Float16* p1) {
  v8h a = *(const v8h*)p0;
  v8h b = *(const v8h*)p1;
  v16h r;
#pragma unroll
  for (int e = 0; e < 8; ++e) { r[e] = a[e]; r[e + 8] = b[e]; }
  return r;
}

// ---------------- Kernel 1: one-time f32->f16 repack into workspace ----------------
// Qh[bh][t][c] = q * 0.125*log2e  (transposed + scaled: WMMA A-layout feed, exp2 domain)
// Kh[bh][t][c] = k                (transposed: B-layout feed for Q@K^T)
// Vh[bh][c][t] = v                (natural:    B-layout feed for P@V)
__global__ __launch_bounds__(256)
void repack_qkv(const float* __restrict__ qkv, _Float16* __restrict__ qh,
                _Float16* __restrict__ kh, _Float16* __restrict__ vh) {
  __shared__ __align__(16) _Float16 qtile[BM * TSTR];
  __shared__ __align__(16) _Float16 ktile[BM * TSTR];
  const int bh = blockIdx.y, t0 = blockIdx.x * BM;
  const int n = bh >> 3, h = bh & 7;
  const size_t hs = (size_t)CHD * TSEQ;
  const float* qp = qkv + (size_t)n * 3 * HEADS * hs + (size_t)h * hs;
  const float* kp = qp + (size_t)HEADS * hs;
  const float* vp = kp + (size_t)HEADS * hs;
  const int tid = threadIdx.x;
  // coalesced float4 reads along t; transpose Q/K via LDS; V converted in place
  for (int i = tid; i < BM * CHD / 4; i += 256) {
    int c = i >> 4, mv = (i & 15) * 4;
    float4 fq = *(const float4*)&qp[(size_t)c * TSEQ + t0 + mv];
    float4 fk = *(const float4*)&kp[(size_t)c * TSEQ + t0 + mv];
    float4 fv = *(const float4*)&vp[(size_t)c * TSEQ + t0 + mv];
    const float s = 0.125f * LOG2E;
    qtile[(mv + 0) * TSTR + c] = (_Float16)(fq.x * s);
    qtile[(mv + 1) * TSTR + c] = (_Float16)(fq.y * s);
    qtile[(mv + 2) * TSTR + c] = (_Float16)(fq.z * s);
    qtile[(mv + 3) * TSTR + c] = (_Float16)(fq.w * s);
    ktile[(mv + 0) * TSTR + c] = (_Float16)fk.x;
    ktile[(mv + 1) * TSTR + c] = (_Float16)fk.y;
    ktile[(mv + 2) * TSTR + c] = (_Float16)fk.z;
    ktile[(mv + 3) * TSTR + c] = (_Float16)fk.w;
    v4h hv = {(_Float16)fv.x, (_Float16)fv.y, (_Float16)fv.z, (_Float16)fv.w};
    *(v4h*)&vh[(size_t)bh * hs + (size_t)c * TSEQ + t0 + mv] = hv;
  }
  __syncthreads();
  // coalesced v8h writes along c
  for (int i = tid; i < BM * CHD / 8; i += 256) {
    int m = i >> 3, c8 = (i & 7) * 8;
    *(v8h*)&qh[(size_t)bh * hs + (size_t)(t0 + m) * CHD + c8] = *(const v8h*)&qtile[m * TSTR + c8];
    *(v8h*)&kh[(size_t)bh * hs + (size_t)(t0 + m) * CHD + c8] = *(const v8h*)&ktile[m * TSTR + c8];
  }
}

// ---------------- Kernel 2: flash attention, all operands direct from L2 ----------------
__global__ __launch_bounds__(128)
void flash_attn_f16_wmma(const _Float16* __restrict__ qh, const _Float16* __restrict__ kh,
                         const _Float16* __restrict__ vh, float* __restrict__ out) {
  __shared__ __align__(16) _Float16 Pt[BM * PSTR];  // per-wave-private P round-trip
  __shared__ __align__(16) float    Ot[CHD * OSTR]; // epilogue transpose

  const int bh = blockIdx.y;
  const int t0 = blockIdx.x * BM;
  const size_t hs = (size_t)CHD * TSEQ;
  const _Float16* qb = qh + (size_t)bh * hs;   // [t][ch]
  const _Float16* kb = kh + (size_t)bh * hs;   // [t][ch]
  const _Float16* vb = vh + (size_t)bh * hs;   // [ch][t]
  float* op = out + (size_t)bh * hs;

  const int tid  = threadIdx.x;
  const int wave = tid >> 5;
  const int lane = tid & 31;
  const int lh   = lane & 15;
  const int hi   = lane >> 4;

  // Q A-fragments (16x32): contiguous in the [t][ch] row, two b128 each.
  v16h aq[2];
  {
    const _Float16* qrow = qb + (size_t)(t0 + wave * 16 + lh) * CHD;
#pragma unroll
    for (int ck = 0; ck < 2; ++ck)
      aq[ck] = ld_frag(qrow + ck * 32 + hi * 8, qrow + ck * 32 + 16 + hi * 8);
  }

  float m_i[8], l_p[8];
  v8f o_acc[4];
#pragma unroll
  for (int r = 0; r < 8; ++r) { m_i[r] = -__builtin_inff(); l_p[r] = 0.f; }
#pragma unroll
  for (int s = 0; s < 4; ++s)
#pragma unroll
    for (int e = 0; e < 8; ++e) o_acc[s][e] = 0.f;

  for (int j = 0; j < TSEQ / BN; ++j) {      // NOTE: zero barriers in this loop
    const int s0 = j * BN;
    if (j + 1 < TSEQ / BN) {                 // probe next tile into cache
      __builtin_prefetch(kb + (size_t)(s0 + BN + lane * 4) * CHD, 0, 0);
      __builtin_prefetch(vb + (size_t)(lane * 2) * TSEQ + s0 + BN, 0, 0);
    }

    // ---- S = Qs @ K^T : B-fragments straight from Kh[t][ch], 16 WMMAs ----
    v8f sc[NSUB];
#pragma unroll
    for (int sub = 0; sub < NSUB; ++sub) {
      v8f acc = {0.f, 0.f, 0.f, 0.f, 0.f, 0.f, 0.f, 0.f};
      const _Float16* krow = kb + (size_t)(s0 + sub * 16 + lh) * CHD;
#pragma unroll
      for (int ck = 0; ck < 2; ++ck) {
        const _Float16* kf = krow + ck * 32 + hi * 8;
        v16h bf = ld_frag(kf, kf + 16);
        acc = __builtin_amdgcn_wmma_f32_16x16x32_f16(false, aq[ck], false, bf,
                                                     (short)0, acc, false, false);
      }
      sc[sub] = acc;
    }

    // ---- Online softmax (exp2 domain); only the MAX crosses lanes ----
#pragma unroll
    for (int r = 0; r < 8; ++r) {
      float mx = sc[0][r];
#pragma unroll
      for (int sub = 1; sub < NSUB; ++sub) mx = fmaxf(mx, sc[sub][r]);
#pragma unroll
      for (int off = 1; off < 16; off <<= 1)
        mx = fmaxf(mx, __shfl_xor(mx, off, 16));
      const float mn    = fmaxf(m_i[r], mx);
      const float alpha = __builtin_amdgcn_exp2f(m_i[r] - mn);
      m_i[r] = mn;
      float rs = 0.f;
#pragma unroll
      for (int sub = 0; sub < NSUB; ++sub) {
        float pv = __builtin_amdgcn_exp2f(sc[sub][r] - mn);
        rs += pv;
        Pt[(wave * 16 + hi * 8 + r) * PSTR + sub * 16 + lh] = (_Float16)pv;
      }
      l_p[r] = l_p[r] * alpha + rs;          // per-lane partial; reduced once at end
#pragma unroll
      for (int sub = 0; sub < 4; ++sub) o_acc[sub][r] *= alpha;
    }

    // ---- O += P @ V : A from per-wave LDS (in-order, no barrier), B from Vh[ch][t] ----
    const _Float16* prow = &Pt[(wave * 16 + lh) * PSTR];
#pragma unroll
    for (int ck = 0; ck < 4; ++ck) {
      v16h af = ld_frag(prow + ck * 32 + hi * 8, prow + ck * 32 + 16 + hi * 8);
#pragma unroll
      for (int sub = 0; sub < 4; ++sub) {
        const _Float16* vf = vb + (size_t)(sub * 16 + lh) * TSEQ + s0 + ck * 32 + hi * 8;
        v16h bf = ld_frag(vf, vf + 16);
        o_acc[sub] = __builtin_amdgcn_wmma_f32_16x16x32_f16(false, af, false, bf,
                                                            (short)0, o_acc[sub],
                                                            false, false);
      }
    }
  }

  // ---- Deferred l reduction, normalize, coalesced store via LDS ----
#pragma unroll
  for (int r = 0; r < 8; ++r) {
#pragma unroll
    for (int off = 1; off < 16; off <<= 1)
      l_p[r] += __shfl_xor(l_p[r], off, 16);
    l_p[r] = 1.0f / l_p[r];
  }
#pragma unroll
  for (int sub = 0; sub < 4; ++sub)
#pragma unroll
    for (int r = 0; r < 8; ++r)
      Ot[(sub * 16 + lh) * OSTR + wave * 16 + hi * 8 + r] = o_acc[sub][r] * l_p[r];
  __syncthreads();
  for (int i = tid; i < CHD * BM; i += 128) {
    int c = i >> 6, m = i & 63;
    op[(size_t)c * TSEQ + t0 + m] = Ot[c * OSTR + m];
  }
}

extern "C" void kernel_launch(void* const* d_in, const int* in_sizes, int n_in,
                              void* d_out, int out_size, void* d_ws, size_t ws_size,
                              hipStream_t stream) {
  const float* qkv = (const float*)d_in[0];
  float* out = (float*)d_out;
  // Workspace: 3 f16 tensors of [32][2048][64] = 8 MB each (24 MB total).
  _Float16* qh = (_Float16*)d_ws;
  _Float16* kh = qh + (size_t)NHTOT * TSEQ * CHD;
  _Float16* vh = kh + (size_t)NHTOT * TSEQ * CHD;
  dim3 grid(TSEQ / BM, NHTOT);
  repack_qkv<<<grid, 256, 0, stream>>>(qkv, qh, kh, vh);
  flash_attn_f16_wmma<<<grid, 128, 0, stream>>>(qh, kh, vh, out);
}
